// GCN_AE_50208167690259
// MI455X (gfx1250) — compile-verified, matrix operand order checked by reading, not run on hardware
//
#include <hip/hip_runtime.h>
#include <hip/hip_bf16.h>
#include <math.h>

#define NN 10000
#define NE 320000

typedef __attribute__((ext_vector_type(16))) __bf16 v16bf;
typedef __attribute__((ext_vector_type(8)))  __bf16 v8bf;
typedef __attribute__((ext_vector_type(8)))  float  v8f;

union FragBF { v16bf v; v8bf h[2]; };

// ---------------- degree / normalization ----------------
__global__ void k_init_deg(int* deg, int n) {
    int i = blockIdx.x * blockDim.x + threadIdx.x;
    if (i < n) deg[i] = 1;                       // self loop
}

__global__ void k_accum_deg(const int* __restrict__ dst, int* deg, int e) {
    int i = blockIdx.x * blockDim.x + threadIdx.x;
    if (i < e) atomicAdd(&deg[dst[i]], 1);
}

__global__ void k_rsqrt_deg(const int* __restrict__ deg, float* __restrict__ dis, int n) {
    int i = blockIdx.x * blockDim.x + threadIdx.x;
    if (i < n) dis[i] = rsqrtf((float)deg[i]);   // deg >= 1 always
}

// ---------------- dtype plumbing ----------------
__global__ void k_cvt_bf16(const float* __restrict__ in, __bf16* __restrict__ out, int n) {
    int i = blockIdx.x * blockDim.x + threadIdx.x;
    if (i < n) out[i] = (__bf16)in[i];
}

// in [rows, cols] f32  ->  out [cols, rows] bf16   (W -> W^T so GEMM reads along K)
__global__ void k_transpose_cvt(const float* __restrict__ in, __bf16* __restrict__ out,
                                int rows, int cols) {
    int idx = blockIdx.x * blockDim.x + threadIdx.x;
    if (idx >= rows * cols) return;
    int c = idx % cols;
    int r = idx / cols;
    out[(size_t)c * rows + r] = (__bf16)in[(size_t)r * cols + c];
}

// ---------------- GCN aggregation ----------------
// agg[i,f] = msg[i,f] * dis[i]^2 + b[f]      (self-loop term + bias)
__global__ void k_self_bias(const float* __restrict__ msg, const float* __restrict__ dis,
                            const float* __restrict__ b, float* __restrict__ agg,
                            int total, int Fmask, int logF) {
    int idx = blockIdx.x * blockDim.x + threadIdx.x;
    if (idx >= total) return;
    int f = idx & Fmask;
    int i = idx >> logF;
    float s = dis[i];
    agg[idx] = msg[idx] * s * s + b[f];
}

// agg[dst,f] += msg[src,f] * dis[src]*dis[dst]    one thread per (edge, feature)
__global__ void k_edge_scatter(const int* __restrict__ src, const int* __restrict__ dst,
                               const float* __restrict__ dis, const float* __restrict__ msg,
                               float* __restrict__ agg, int total, int F, int Fmask, int logF) {
    int idx = blockIdx.x * blockDim.x + threadIdx.x;
    if (idx >= total) return;
    int f = idx & Fmask;
    int e = idx >> logF;
    int s = src[e], d = dst[e];
    float nrm = dis[s] * dis[d];
    atomicAdd(&agg[(size_t)d * F + f], msg[(size_t)s * F + f] * nrm);
}

// ---------------- WMMA GEMM:  C[M,N] = A[M,K] * B[N,K]^T  (both row-major bf16) ----------------
// One wave per 16x16 output tile. K multiple of 32. Optional fused sigmoid epilogue.
// A-fragment (16-bit 16x32, ISA 7.12.2): lane L holds row M=L&15; K-half = (L>>4)*8;
// elements 0..7 -> K = khalf*8..+7, elements 8..15 -> K = 16+khalf*8..+7  => two 16B loads.
// B-fragment symmetric with N=L&15 (B column n = row n of the NT-stored operand).
template <bool SIGMOID>
__global__ void k_gemm_nt_bf16(const __bf16* __restrict__ A, const __bf16* __restrict__ B,
                               float* __restrict__ C, int M, int N, int K) {
    int wid  = blockIdx.x * (blockDim.x >> 5) + (threadIdx.x >> 5);
    int lane = threadIdx.x & 31;
    int tn = N >> 4;
    if (wid >= (M >> 4) * tn) return;            // wave-uniform: EXEC stays all-1 for WMMA
    int tm = wid / tn;
    int tj = wid - tm * tn;

    int rc    = lane & 15;                       // row of A / row of B (= output column)
    int khalf = lane >> 4;                       // which K-half this lane holds

    const __bf16* pa = A + (size_t)(tm * 16 + rc) * K + khalf * 8;
    const __bf16* pb = B + (size_t)(tj * 16 + rc) * K + khalf * 8;

    v8f c = {};
    for (int k0 = 0; k0 < K; k0 += 32) {
        FragBF a, b;
        a.h[0] = *(const v8bf*)(pa + k0);
        a.h[1] = *(const v8bf*)(pa + k0 + 16);
        b.h[0] = *(const v8bf*)(pb + k0);
        b.h[1] = *(const v8bf*)(pb + k0 + 16);
        c = __builtin_amdgcn_wmma_f32_16x16x32_bf16(false, a.v, false, b.v,
                                                    (short)0, c, false, false);
    }

    // C layout: VGPR v -> M = v + khalf*8 ; N = lane&15
    size_t base = (size_t)(tm * 16 + khalf * 8) * N + (size_t)tj * 16 + rc;
#pragma unroll
    for (int v = 0; v < 8; ++v) {
        float val = c[v];
        if (SIGMOID) val = 1.0f / (1.0f + __expf(-val));
        C[base + (size_t)v * N] = val;
    }
}

// ---------------- host orchestration ----------------
static inline size_t alignup(size_t x) { return (x + 255) & ~(size_t)255; }

extern "C" void kernel_launch(void* const* d_in, const int* in_sizes, int n_in,
                              void* d_out, int out_size, void* d_ws, size_t ws_size,
                              hipStream_t stream) {
    const float* X   = (const float*)d_in[0];          // [10000,512]
    const int*   ei  = (const int*)d_in[1];            // [2,320000]
    const float* W1  = (const float*)d_in[2];          // [512,256]
    const float* b1  = (const float*)d_in[3];          // [256]
    const float* W2  = (const float*)d_in[4];          // [256,64]
    const float* b2  = (const float*)d_in[5];          // [64]
    float*       Y   = (float*)d_out;                  // [10000,10000]

    const int* src = ei;
    const int* dst = ei + NE;

    // workspace carve
    char* p = (char*)d_ws;
    int*    degi = (int*)p;     p += alignup((size_t)NN * 4);
    float*  dis  = (float*)p;   p += alignup((size_t)NN * 4);
    __bf16* Xh   = (__bf16*)p;  p += alignup((size_t)NN * 512 * 2);
    __bf16* W1t  = (__bf16*)p;  p += alignup((size_t)256 * 512 * 2);
    __bf16* W2t  = (__bf16*)p;  p += alignup((size_t)64 * 256 * 2);
    float*  XW1  = (float*)p;   p += alignup((size_t)NN * 256 * 4);   // X @ W1
    float*  H1   = (float*)p;   p += alignup((size_t)NN * 256 * 4);   // aggregated + bias
    __bf16* H1h  = (__bf16*)p;  p += alignup((size_t)NN * 256 * 2);
    float*  HW2  = (float*)p;   p += alignup((size_t)NN * 64 * 4);    // H1 @ W2
    float*  Z    = (float*)p;   p += alignup((size_t)NN * 64 * 4);
    __bf16* Zh   = (__bf16*)p;  p += alignup((size_t)NN * 64 * 2);

    const int T = 256;
    // 1) symmetric normalization  D^{-1/2}
    k_init_deg <<<(NN + T - 1) / T, T, 0, stream>>>(degi, NN);
    k_accum_deg<<<(NE + T - 1) / T, T, 0, stream>>>(dst, degi, NE);
    k_rsqrt_deg<<<(NN + T - 1) / T, T, 0, stream>>>(degi, dis, NN);

    // 2) bf16 operands
    k_cvt_bf16     <<<(NN * 512 + T - 1) / T, T, 0, stream>>>(X, Xh, NN * 512);
    k_transpose_cvt<<<(512 * 256 + T - 1) / T, T, 0, stream>>>(W1, W1t, 512, 256);
    k_transpose_cvt<<<(256 * 64  + T - 1) / T, T, 0, stream>>>(W2, W2t, 256, 64);

    // 3) layer 1: GEMM first (aggregation is linear), then gather/scatter in 256-dim
    {
        int tiles = (NN / 16) * (256 / 16);
        k_gemm_nt_bf16<false><<<(tiles + 3) / 4, 128, 0, stream>>>(Xh, W1t, XW1, NN, 256, 512);
    }
    k_self_bias   <<<(NN * 256 + T - 1) / T, T, 0, stream>>>(XW1, dis, b1, H1, NN * 256, 255, 8);
    k_edge_scatter<<<(NE * 256 + T - 1) / T, T, 0, stream>>>(src, dst, dis, XW1, H1,
                                                             NE * 256, 256, 255, 8);

    // 4) layer 2
    k_cvt_bf16<<<(NN * 256 + T - 1) / T, T, 0, stream>>>(H1, H1h, NN * 256);
    {
        int tiles = (NN / 16) * (64 / 16);
        k_gemm_nt_bf16<false><<<(tiles + 3) / 4, 128, 0, stream>>>(H1h, W2t, HW2, NN, 64, 256);
    }
    k_self_bias   <<<(NN * 64 + T - 1) / T, T, 0, stream>>>(HW2, dis, b2, Z, NN * 64, 63, 6);
    k_edge_scatter<<<(NE * 64 + T - 1) / T, T, 0, stream>>>(src, dst, dis, HW2, Z,
                                                            NE * 64, 64, 63, 6);

    // 5) decode: Y = sigmoid(Z @ Z^T)  — Zh (1.28 MB) resides in L2; store-bound
    k_cvt_bf16<<<(NN * 64 + T - 1) / T, T, 0, stream>>>(Z, Zh, NN * 64);
    {
        int tiles = (NN / 16) * (NN / 16);           // 625 * 625
        k_gemm_nt_bf16<true><<<(tiles + 3) / 4, 128, 0, stream>>>(Zh, Zh, Y, NN, NN, 64);
    }
}
